// GraphAttentionLayer_3410204033346
// MI455X (gfx1250) — matrix operand order repaired and checked by reading, compiled, not run on hardware
//
#include <hip/hip_runtime.h>
#include <hip/hip_bf16.h>

typedef __attribute__((ext_vector_type(2)))  float  v2f;
typedef __attribute__((ext_vector_type(8)))  float  v8f;
typedef __attribute__((ext_vector_type(16))) __bf16 v16bf;

#define N_NODES 8192
#define FIN     256
#define FOUT    128
#define GAT_ALPHA 0.2f

// ---------------------------------------------------------------------------
// Kernel A: h = input @ W  (f32 WMMA 16x16x4), also emit bf16 transposed copy
// block = 256 thr = 8 waves; block owns rows x0..x0+15, wave w owns cols 16w..16w+15
// ---------------------------------------------------------------------------
__global__ __launch_bounds__(256) void k_gemm_h(const float* __restrict__ in,
                                                const float* __restrict__ W,
                                                float* __restrict__ h,
                                                __bf16* __restrict__ hT)
{
    const int lane = threadIdx.x & 31;
    const int wave = threadIdx.x >> 5;
    const int x0   = blockIdx.x * 16;
    const int f0   = wave * 16;
    const int half = lane >> 4;    // K pair selector for f32 A/B fragments
    const int l15  = lane & 15;

    v8f acc = {};
    const float* arow = in + (size_t)(x0 + l15) * FIN + half * 2;
    const float* bcol = W + (size_t)(half * 2) * FOUT + f0 + l15;
    for (int k0 = 0; k0 < FIN; k0 += 4) {
        v2f a = *(const v2f*)(arow + k0);               // A(m, k0+2h .. +1)
        v2f b;
        b.x = bcol[(size_t)k0 * FOUT];                  // B(k0+2h,   n)
        b.y = bcol[(size_t)(k0 + 1) * FOUT];            // B(k0+2h+1, n)
        acc = __builtin_amdgcn_wmma_f32_16x16x4_f32(false, a, false, b,
                                                    (short)0, acc, false, false);
    }
    // D(m,n): lane = n + 16*(m>=8), vgpr r = m&7
#pragma unroll
    for (int r = 0; r < 8; ++r) {
        int   m = r + 8 * half;
        float v = acc[r];
        h [(size_t)(x0 + m) * FOUT + f0 + l15] = v;
        hT[(size_t)(f0 + l15) * N_NODES + x0 + m] = (__bf16)v;
    }
}

// ---------------------------------------------------------------------------
// Kernel B: per node j: s1[j]=h[j]·c1, s2[j]=h[j]·c2, jnz[j]=(sum(h[j])!=0)
// ---------------------------------------------------------------------------
__global__ __launch_bounds__(128) void k_rowfeat(const float* __restrict__ h,
                                                 const float* __restrict__ c1,
                                                 const float* __restrict__ c2,
                                                 float* __restrict__ s1,
                                                 float* __restrict__ s2,
                                                 unsigned char* __restrict__ jnz)
{
    __shared__ float r0[128], r1[128], r2[128];
    const int j = blockIdx.x;
    const int t = threadIdx.x;
    float v = h[(size_t)j * FOUT + t];
    r0[t] = v; r1[t] = v * c1[t]; r2[t] = v * c2[t];
    __syncthreads();
    for (int s = 64; s > 0; s >>= 1) {
        if (t < s) { r0[t] += r0[t + s]; r1[t] += r1[t + s]; r2[t] += r2[t + s]; }
        __syncthreads();
    }
    if (t == 0) {
        s1[j] = r1[0];
        s2[j] = r2[0];
        jnz[j] = (r0[0] != 0.0f) ? 1 : 0;
    }
}

// ---------------------------------------------------------------------------
// Kernel C: compress adj (256 MB) into validity bitmask (8 MB)
// one thread -> one 32-bit word (32 neighbors, 128B contiguous adj read)
// ---------------------------------------------------------------------------
__global__ __launch_bounds__(256) void k_mask(const int* __restrict__ adj,
                                              const unsigned char* __restrict__ jnz,
                                              unsigned int* __restrict__ vb)
{
    const unsigned int w = blockIdx.x * 256u + threadIdx.x; // word index
    const int x    = (int)(w >> 8);       // 256 words per row
    const int wrow = (int)(w & 255u);
    const int4* ap = (const int4*)(adj + (size_t)x * N_NODES + wrow * 32);
    const unsigned char* jp = jnz + wrow * 32;
    unsigned int bits = 0u;
#pragma unroll
    for (int i = 0; i < 8; ++i) {
        int4 a = ap[i];
        if (a.x > 0 && jp[i * 4 + 0]) bits |= 1u << (i * 4 + 0);
        if (a.y > 0 && jp[i * 4 + 1]) bits |= 1u << (i * 4 + 1);
        if (a.z > 0 && jp[i * 4 + 2]) bits |= 1u << (i * 4 + 2);
        if (a.w > 0 && jp[i * 4 + 3]) bits |= 1u << (i * 4 + 3);
    }
    vb[w] = bits;
}

// ---------------------------------------------------------------------------
// Kernel D: per row x: softmax stats (max over valid e, 1/sum exp)
// ---------------------------------------------------------------------------
__global__ __launch_bounds__(256) void k_rowsoftmax(const unsigned int* __restrict__ vb,
                                                    const float* __restrict__ s1,
                                                    const float* __restrict__ s2,
                                                    float* __restrict__ rowmax,
                                                    float* __restrict__ invZ)
{
    __shared__ float red[256];
    __shared__ int   cred[256];
    const int x = blockIdx.x;
    const int t = threadIdx.x;
    const float s1x = s1[x];
    const unsigned int* vrow = vb + (size_t)x * 256;

    float mx = -3.0e38f;
    for (int j = t; j < N_NODES; j += 256) {
        if ((vrow[j >> 5] >> (j & 31)) & 1u) {
            float e = s1x + s2[j];
            e = e > 0.f ? e : GAT_ALPHA * e;
            mx = fmaxf(mx, e);
        }
    }
    red[t] = mx; __syncthreads();
    for (int s = 128; s > 0; s >>= 1) {
        if (t < s) red[t] = fmaxf(red[t], red[t + s]);
        __syncthreads();
    }
    mx = red[0]; __syncthreads();

    float Z = 0.f; int cnt = 0;
    for (int j = t; j < N_NODES; j += 256) {
        if ((vrow[j >> 5] >> (j & 31)) & 1u) {
            float e = s1x + s2[j];
            e = e > 0.f ? e : GAT_ALPHA * e;
            Z += __expf(e - mx); cnt++;
        }
    }
    red[t] = Z; cred[t] = cnt; __syncthreads();
    for (int s = 128; s > 0; s >>= 1) {
        if (t < s) { red[t] += red[t + s]; cred[t] += cred[t + s]; }
        __syncthreads();
    }
    if (t == 0) {
        bool ok = (cred[0] > 0) && (red[0] > 0.f);
        rowmax[x] = ok ? mx : 0.f;
        invZ[x]   = ok ? 1.f / red[0] : 0.f;   // no-neighbor rows -> all-zero weights
    }
}

// ---------------------------------------------------------------------------
// Kernel E: out = elu(softmax(e) @ h), bf16 WMMA 16x16x32
// block = 256 thr = 8 waves; rows x0..x0+15; wave w -> features 16w..16w+15
// K=128 per barrier pair: 4 A-chunks staged in LDS, 4 WMMAs back-to-back
// ---------------------------------------------------------------------------
__global__ __launch_bounds__(256) void k_attn_gemm(const unsigned int* __restrict__ vb,
                                                   const float* __restrict__ s1,
                                                   const float* __restrict__ s2,
                                                   const float* __restrict__ rowmax,
                                                   const float* __restrict__ invZ,
                                                   const __bf16* __restrict__ hT,
                                                   float* __restrict__ out)
{
    __shared__ __bf16 Afrag[4 * 512];       // 4 K-chunks, fragment order [lane][16]
    __shared__ float ls1[16], lmx[16], liz[16];

    const int t    = threadIdx.x;
    const int lane = t & 31;
    const int wave = t >> 5;
    const int x0   = blockIdx.x * 16;

    if (t < 16) { ls1[t] = s1[x0 + t]; lmx[t] = rowmax[x0 + t]; liz[t] = invZ[x0 + t]; }
    __syncthreads();

    // A-builder mapping: thread -> (fragment lane fl, value pair i0,i0+1) per chunk
    const int fl    = t >> 3;          // 0..31
    const int i0    = (t & 7) * 2;     // 0,2,..,14
    const int am    = fl & 15;         // A row within tile
    const int ahalf = fl >> 4;
    const float as1 = ls1[am], amx = lmx[am], aiz = liz[am];
    const unsigned int* vrow = vb + (size_t)(x0 + am) * 256;

    // B consumer mapping: lane holds column f, K in [16*half, 16*half+16)
    const int half = lane >> 4;
    const int l15  = lane & 15;
    const int f    = wave * 16 + l15;
    const __bf16* bptr = hT + (size_t)f * N_NODES + half * 16;

    v8f acc = {};
    for (int j0 = 0; j0 < N_NODES; j0 += 128) {
        // software prefetch of the next B stripe into the WGP cache
        if (j0 + 128 < N_NODES)
            __builtin_prefetch(bptr + j0 + 128, 0, 0);   // global_prefetch_b8

        // build 4x(16x32) bf16 attention fragments; exp once, shared by 8 waves
#pragma unroll
        for (int c = 0; c < 4; ++c) {
#pragma unroll
            for (int u = 0; u < 2; ++u) {
                int i  = i0 + u;
                int kl = ahalf * 16 + i;
                int j  = j0 + c * 32 + kl;
                float a = 0.f;
                if ((vrow[j >> 5] >> (j & 31)) & 1u) {
                    float e = as1 + s2[j];
                    e = e > 0.f ? e : GAT_ALPHA * e;
                    a = __expf(e - amx) * aiz;
                }
                Afrag[c * 512 + fl * 16 + i] = (__bf16)a;
            }
        }
        __syncthreads();
        v16bf A0 = *(const v16bf*)(Afrag +        lane * 16);
        v16bf A1 = *(const v16bf*)(Afrag +  512 + lane * 16);
        v16bf A2 = *(const v16bf*)(Afrag + 1024 + lane * 16);
        v16bf A3 = *(const v16bf*)(Afrag + 1536 + lane * 16);
        v16bf B0 = *(const v16bf*)(bptr + j0);
        v16bf B1 = *(const v16bf*)(bptr + j0 + 32);
        v16bf B2 = *(const v16bf*)(bptr + j0 + 64);
        v16bf B3 = *(const v16bf*)(bptr + j0 + 96);
        acc = __builtin_amdgcn_wmma_f32_16x16x32_bf16(false, A0, false, B0,
                                                      (short)0, acc, false, false);
        acc = __builtin_amdgcn_wmma_f32_16x16x32_bf16(false, A1, false, B1,
                                                      (short)0, acc, false, false);
        acc = __builtin_amdgcn_wmma_f32_16x16x32_bf16(false, A2, false, B2,
                                                      (short)0, acc, false, false);
        acc = __builtin_amdgcn_wmma_f32_16x16x32_bf16(false, A3, false, B3,
                                                      (short)0, acc, false, false);
        __syncthreads();
    }

    // epilogue: ELU (alpha=1) and store
#pragma unroll
    for (int r = 0; r < 8; ++r) {
        int   m = r + 8 * half;
        float v = acc[r];
        v = v > 0.f ? v : (__expf(v) - 1.f);
        out[(size_t)(x0 + m) * FOUT + wave * 16 + l15] = v;
    }
}

// ---------------------------------------------------------------------------
extern "C" void kernel_launch(void* const* d_in, const int* in_sizes, int n_in,
                              void* d_out, int out_size, void* d_ws, size_t ws_size,
                              hipStream_t stream)
{
    (void)in_sizes; (void)n_in; (void)out_size; (void)ws_size;
    const float* input = (const float*)d_in[0];   // [8192,256]
    const float* W     = (const float*)d_in[1];   // [256,128]
    const float* c1    = (const float*)d_in[2];   // [128,1]
    const float* c2    = (const float*)d_in[3];   // [128,1]
    const int*   adj   = (const int*)  d_in[4];   // [8192,8192]
    // d_in[5] = M (scalar, == 8192 for this problem); shapes hardcoded.

    char* ws = (char*)d_ws;
    size_t off = 0;
    float*  h      = (float*) (ws + off); off += (size_t)N_NODES * FOUT * 4;   // 4 MB
    __bf16* hT     = (__bf16*)(ws + off); off += (size_t)N_NODES * FOUT * 2;   // 2 MB
    float*  s1     = (float*) (ws + off); off += N_NODES * 4;
    float*  s2     = (float*) (ws + off); off += N_NODES * 4;
    float*  rmax   = (float*) (ws + off); off += N_NODES * 4;
    float*  iZ     = (float*) (ws + off); off += N_NODES * 4;
    unsigned char* jnz = (unsigned char*)(ws + off); off += 32768;             // padded
    unsigned int* vbits = (unsigned int*)(ws + off); off += (size_t)N_NODES * 256 * 4; // 8 MB

    k_gemm_h    <<<N_NODES / 16, 256, 0, stream>>>(input, W, h, hT);
    k_rowfeat   <<<N_NODES,      128, 0, stream>>>(h, c1, c2, s1, s2, jnz);
    k_mask      <<<N_NODES,      256, 0, stream>>>(adj, jnz, vbits);
    k_rowsoftmax<<<N_NODES,      256, 0, stream>>>(vbits, s1, s2, rmax, iZ);
    k_attn_gemm <<<N_NODES / 16, 256, 0, stream>>>(vbits, s1, s2, rmax, iZ, hT,
                                                   (float*)d_out);
}
